// LMModel_LSTM_57870389346623
// MI455X (gfx1250) — matrix operand order repaired and checked by reading, not compiled, not run on hardware
//
#include <hip/hip_runtime.h>

// ---------------------------------------------------------------------------
// LSTM LM for MI455X (gfx1250, wave32, WMMA).
// Layer-by-layer decomposition: input-weight contribution of all 128 steps is
// one big bf16-WMMA GEMM; only h@U^T (32x2048x512) stays sequential.
// GEMM uses double-buffered LDS staging via GLOBAL_LOAD_ASYNC_TO_LDS_B128
// (ASYNCcnt) when the builtin exists, overlapping DMA with v_wmma.
// ---------------------------------------------------------------------------

#define SEQL  128
#define BATCH 32
#define DIM   512
#define HID   512
#define NVOC  32000
#define ROWS  (SEQL * BATCH)   // 4096
#define G4    (4 * HID)        // 2048

#if defined(__has_builtin)
#  if __has_builtin(__builtin_amdgcn_global_load_async_to_lds_b128)
#    define HAVE_ASYNC_LDS 1
#  endif
#endif

typedef __attribute__((ext_vector_type(16))) __bf16 v16bf;
typedef __attribute__((ext_vector_type(8)))  float  v8f;
typedef int v4i __attribute__((vector_size(16)));
typedef __attribute__((address_space(1))) v4i* gv4p;   // global int4*
typedef __attribute__((address_space(3))) v4i* lv4p;   // LDS int4*

union FragBF { v16bf v; uint4 q[2]; };

__device__ __forceinline__ __bf16 f2bf(float f) {
  union { float f; unsigned u; } in; in.f = f;
  unsigned r = (in.u + 0x7FFFu + ((in.u >> 16) & 1u)) >> 16;   // RNE
  union { unsigned short s; __bf16 b; } o; o.s = (unsigned short)r;
  return o.b;
}
__device__ __forceinline__ float sigmoidf(float x) { return 1.0f / (1.0f + __expf(-x)); }

__device__ __forceinline__ void wait_async0() {
#if defined(__has_builtin) && __has_builtin(__builtin_amdgcn_s_wait_asynccnt)
  __builtin_amdgcn_s_wait_asynccnt(0);
#else
  asm volatile("s_wait_asynccnt 0x0" ::: "memory");
#endif
}

// --------------------------- fp32 -> bf16 ----------------------------------
__global__ void k_f32_to_bf16(const float* __restrict__ src, __bf16* __restrict__ dst, int n) {
  int i = blockIdx.x * blockDim.x + threadIdx.x;
  if (i < n) dst[i] = f2bf(src[i]);
}

// --------------------------- embedding gather ------------------------------
__global__ void k_embed(const int* __restrict__ tokens, const float* __restrict__ emb,
                        __bf16* __restrict__ X) {
  int idx = blockIdx.x * 256 + threadIdx.x;     // ROWS*DIM threads
  int r = idx >> 9;
  int d = idx & 511;
  X[idx] = f2bf(emb[(size_t)tokens[r] * DIM + d]);
}

// --------------------------- tiled bf16 WMMA GEMM --------------------------
// C[M,N] = A[M,K] @ B[N,K]^T (+bias).  Block tile 64x64, 4 waves; each wave:
// 16 M-rows x 4 N-subtiles (4x v8f accumulators).  32-wide K slabs staged
// through double-buffered LDS; async DMA overlapped with WMMA.
// Requires M,N % 64 == 0, K % 32 == 0.
__global__ void __launch_bounds__(128)
k_gemm_bf16(const __bf16* __restrict__ A, const __bf16* __restrict__ B,
            float* __restrict__ C, const float* __restrict__ bias, int N, int K) {
  __shared__ __bf16 At[2][64][32];
  __shared__ __bf16 Bt[2][64][32];

  const int tid  = threadIdx.x;
  const int lane = tid & 31;
  const int wv   = tid >> 5;          // wave 0..3 -> M rows [16wv,16wv+16)
  const int lm   = lane & 15;
  const int Mb   = blockIdx.y * 64;
  const int Nb   = blockIdx.x * 64;

  // staging map: each thread owns one 32-byte chunk per tile per slab
  const int crow = tid >> 1;          // 0..63
  const int ce   = (tid & 1) * 16;    // element offset in 32-elem row

  const __bf16* gA = A + (size_t)(Mb + crow) * K + ce;   // advance +32/slab
  const __bf16* gB = B + (size_t)(Nb + crow) * K + ce;

  const int ab = (lane < 16) ? 0 : 16;  // A fragment k base
  const int bb = (lane < 16) ? 0 : 16;  // B fragment k base

  v8f acc0 = {}, acc1 = {}, acc2 = {}, acc3 = {};
  const int nk = K >> 5;

  auto stage = [&](int bufi) {
#if HAVE_ASYNC_LDS
    __builtin_amdgcn_global_load_async_to_lds_b128(
        (gv4p)(void*)(const void*)gA,       (lv4p)(void*)&At[bufi][crow][ce],     0, 0);
    __builtin_amdgcn_global_load_async_to_lds_b128(
        (gv4p)(void*)(const void*)(gA + 8), (lv4p)(void*)&At[bufi][crow][ce + 8], 0, 0);
    __builtin_amdgcn_global_load_async_to_lds_b128(
        (gv4p)(void*)(const void*)gB,       (lv4p)(void*)&Bt[bufi][crow][ce],     0, 0);
    __builtin_amdgcn_global_load_async_to_lds_b128(
        (gv4p)(void*)(const void*)(gB + 8), (lv4p)(void*)&Bt[bufi][crow][ce + 8], 0, 0);
#else
    uint4 a0 = *(const uint4*)gA, a1 = *(const uint4*)(gA + 8);
    uint4 b0 = *(const uint4*)gB, b1 = *(const uint4*)(gB + 8);
    *(uint4*)&At[bufi][crow][ce]     = a0;
    *(uint4*)&At[bufi][crow][ce + 8] = a1;
    *(uint4*)&Bt[bufi][crow][ce]     = b0;
    *(uint4*)&Bt[bufi][crow][ce + 8] = b1;
#endif
    gA += 32; gB += 32;
  };

  stage(0);
  int buf = 0;
  for (int kt = 0; kt < nk; ++kt) {
#if HAVE_ASYNC_LDS
    wait_async0();                 // my slab-kt DMA complete
#endif
    __syncthreads();               // everyone's slab-kt staged & visible
    if (kt + 1 < nk) stage(buf ^ 1);
    if (kt + 2 < nk) {             // warm L2 for the slab after next
      __builtin_prefetch(gA, 0, 1);
      __builtin_prefetch(gB, 0, 1);
    }

    FragBF a;
    a.q[0] = *(const uint4*)&At[buf][wv * 16 + lm][ab];
    a.q[1] = *(const uint4*)&At[buf][wv * 16 + lm][ab + 8];

    FragBF b0, b1, b2, b3;
    b0.q[0] = *(const uint4*)&Bt[buf][ 0 + lm][bb]; b0.q[1] = *(const uint4*)&Bt[buf][ 0 + lm][bb + 8];
    b1.q[0] = *(const uint4*)&Bt[buf][16 + lm][bb]; b1.q[1] = *(const uint4*)&Bt[buf][16 + lm][bb + 8];
    b2.q[0] = *(const uint4*)&Bt[buf][32 + lm][bb]; b2.q[1] = *(const uint4*)&Bt[buf][32 + lm][bb + 8];
    b3.q[0] = *(const uint4*)&Bt[buf][48 + lm][bb]; b3.q[1] = *(const uint4*)&Bt[buf][48 + lm][bb + 8];

    acc0 = __builtin_amdgcn_wmma_f32_16x16x32_bf16(false, a.v, false, b0.v, (short)0, acc0, false, false);
    acc1 = __builtin_amdgcn_wmma_f32_16x16x32_bf16(false, a.v, false, b1.v, (short)0, acc1, false, false);
    acc2 = __builtin_amdgcn_wmma_f32_16x16x32_bf16(false, a.v, false, b2.v, (short)0, acc2, false, false);
    acc3 = __builtin_amdgcn_wmma_f32_16x16x32_bf16(false, a.v, false, b3.v, (short)0, acc3, false, false);

    buf ^= 1;
    __syncthreads();               // protect buf reuse
  }

  float bv0 = 0.f, bv1 = 0.f, bv2 = 0.f, bv3 = 0.f;
  if (bias) {
    bv0 = bias[Nb +  0 + lm]; bv1 = bias[Nb + 16 + lm];
    bv2 = bias[Nb + 32 + lm]; bv3 = bias[Nb + 48 + lm];
  }
  const int roff = (lane >> 4) << 3;   // C/D layout: row = r + 8*(lane>=16)
  #pragma unroll
  for (int r = 0; r < 8; ++r) {
    size_t row = (size_t)(Mb + wv * 16 + r + roff);
    float* crow_p = &C[row * (size_t)N + Nb];
    crow_p[ 0 + lm] = acc0[r] + bv0;
    crow_p[16 + lm] = acc1[r] + bv1;
    crow_p[32 + lm] = acc2[r] + bv2;
    crow_p[48 + lm] = acc3[r] + bv3;
  }
}

// --------------------------- fused recurrent step --------------------------
// gates[32,4H] = gpre_t + hprev @ U^T (+b); then LSTM cell for this block's
// 16 hidden columns.  grid = HID/16 blocks, block = 64 threads (2 waves).
__global__ void __launch_bounds__(64)
k_lstm_step(const __bf16* __restrict__ hprev, const __bf16* __restrict__ U,
            const float* __restrict__ gpre_t, const float* __restrict__ bias,
            float* __restrict__ cst, __bf16* __restrict__ hnext,
            __bf16* __restrict__ hist_t) {
  const int tid  = threadIdx.x;
  const int lane = tid & 31;
  const int wv   = tid >> 5;            // batch rows [16wv, 16wv+16)
  const int lm   = lane & 15;
  const int n0   = blockIdx.x * 16;     // hidden-column tile
  const int ab   = (lane < 16) ? 0 : 16;
  const int bb   = (lane < 16) ? 0 : 16;

  v8f acc0 = {}, acc1 = {}, acc2 = {}, acc3 = {};

  const __bf16* pa = hprev + (size_t)(wv * 16 + lm) * HID;        // +32/iter
  const __bf16* p0 = U + (size_t)(0 * HID + n0 + lm) * HID + bb;  // gate i
  const __bf16* p1 = U + (size_t)(1 * HID + n0 + lm) * HID + bb;  // gate f
  const __bf16* p2 = U + (size_t)(2 * HID + n0 + lm) * HID + bb;  // gate g
  const __bf16* p3 = U + (size_t)(3 * HID + n0 + lm) * HID + bb;  // gate o

  for (int kt = 0; kt < HID / 32; ++kt) {
    FragBF a;
    a.q[0] = *(const uint4*)(pa + ab);
    a.q[1] = *(const uint4*)(pa + ab + 8);
    FragBF b;
    b.q[0] = *(const uint4*)p0; b.q[1] = *(const uint4*)(p0 + 8);
    acc0 = __builtin_amdgcn_wmma_f32_16x16x32_bf16(false, a.v, false, b.v, (short)0, acc0, false, false);
    b.q[0] = *(const uint4*)p1; b.q[1] = *(const uint4*)(p1 + 8);
    acc1 = __builtin_amdgcn_wmma_f32_16x16x32_bf16(false, a.v, false, b.v, (short)0, acc1, false, false);
    b.q[0] = *(const uint4*)p2; b.q[1] = *(const uint4*)(p2 + 8);
    acc2 = __builtin_amdgcn_wmma_f32_16x16x32_bf16(false, a.v, false, b.v, (short)0, acc2, false, false);
    b.q[0] = *(const uint4*)p3; b.q[1] = *(const uint4*)(p3 + 8);
    acc3 = __builtin_amdgcn_wmma_f32_16x16x32_bf16(false, a.v, false, b.v, (short)0, acc3, false, false);
    pa += 32; p0 += 32; p1 += 32; p2 += 32; p3 += 32;
  }

  const int roff = (lane >> 4) << 3;
  const int n = n0 + lm;
  #pragma unroll
  for (int r = 0; r < 8; ++r) {
    int m = wv * 16 + r + roff;                       // batch row
    const float* gp = &gpre_t[(size_t)m * G4];
    float gi = acc0[r] + gp[0 * HID + n] + bias[0 * HID + n];
    float gf = acc1[r] + gp[1 * HID + n] + bias[1 * HID + n];
    float gg = acc2[r] + gp[2 * HID + n] + bias[2 * HID + n];
    float go = acc3[r] + gp[3 * HID + n] + bias[3 * HID + n];
    float iv = sigmoidf(gi), fv = sigmoidf(gf);
    float gv = tanhf(gg),    ov = sigmoidf(go);
    size_t ci = (size_t)m * HID + n;
    float cn = fv * cst[ci] + iv * gv;
    cst[ci] = cn;
    __bf16 hb = f2bf(ov * tanhf(cn));
    hnext[ci]  = hb;
    hist_t[ci] = hb;
  }
}

// ---------------------------------------------------------------------------
extern "C" void kernel_launch(void* const* d_in, const int* in_sizes, int n_in,
                              void* d_out, int out_size, void* d_ws, size_t ws_size,
                              hipStream_t stream) {
  const int*   tokens = (const int*)  d_in[0];
  const float* emb    = (const float*)d_in[1];
  const float* W0     = (const float*)d_in[2];
  const float* W1     = (const float*)d_in[3];
  const float* U0     = (const float*)d_in[4];
  const float* U1     = (const float*)d_in[5];
  const float* bvec   = (const float*)d_in[6];
  const float* dec_w  = (const float*)d_in[7];
  const float* dec_b  = (const float*)d_in[8];
  float* out = (float*)d_out;

  char*  ws  = (char*)d_ws;
  size_t off = 0;
  auto alloc = [&](size_t bytes) { void* p = ws + off; off = (off + bytes + 255) & ~255ull; return p; };

  __bf16* W0b  = (__bf16*)alloc((size_t)G4 * DIM * 2);
  __bf16* W1b  = (__bf16*)alloc((size_t)G4 * DIM * 2);
  __bf16* U0b  = (__bf16*)alloc((size_t)G4 * HID * 2);
  __bf16* U1b  = (__bf16*)alloc((size_t)G4 * HID * 2);
  __bf16* Db   = (__bf16*)alloc((size_t)NVOC * HID * 2);
  __bf16* Xb   = (__bf16*)alloc((size_t)ROWS * DIM * 2);
  __bf16* H0   = (__bf16*)alloc((size_t)ROWS * HID * 2);      // layer-0 outs
  __bf16* H1   = (__bf16*)alloc((size_t)ROWS * HID * 2);      // layer-1 outs
  __bf16* hpp  = (__bf16*)alloc((size_t)2 * BATCH * HID * 2); // h ping-pong
  float*  Gpre = (float*) alloc((size_t)ROWS * G4 * 4);       // x@W^T all t
  float*  cst  = (float*) alloc((size_t)BATCH * HID * 4);     // cell state
  if (off > ws_size) return;

  // 1) weights -> bf16 (recomputed every call: no cached state)
  int nw = G4 * DIM;
  k_f32_to_bf16<<<(nw + 255) / 256, 256, 0, stream>>>(W0, W0b, nw);
  k_f32_to_bf16<<<(nw + 255) / 256, 256, 0, stream>>>(W1, W1b, nw);
  k_f32_to_bf16<<<(nw + 255) / 256, 256, 0, stream>>>(U0, U0b, nw);
  k_f32_to_bf16<<<(nw + 255) / 256, 256, 0, stream>>>(U1, U1b, nw);
  int nd = NVOC * HID;
  k_f32_to_bf16<<<(nd + 255) / 256, 256, 0, stream>>>(dec_w, Db, nd);

  // 2) embedding gather
  k_embed<<<(ROWS * DIM) / 256, 256, 0, stream>>>(tokens, emb, Xb);

  // 3) layer 0
  k_gemm_bf16<<<dim3(G4 / 64, ROWS / 64), 128, 0, stream>>>(Xb, W0b, Gpre, nullptr, G4, DIM);
  (void)hipMemsetAsync(cst, 0, (size_t)BATCH * HID * 4, stream);
  (void)hipMemsetAsync(hpp, 0, (size_t)2 * BATCH * HID * 2, stream);
  for (int t = 0; t < SEQL; ++t) {
    const __bf16* hin = hpp + (size_t)(t & 1) * BATCH * HID;
    __bf16* hout      = hpp + (size_t)((t + 1) & 1) * BATCH * HID;
    k_lstm_step<<<HID / 16, 64, 0, stream>>>(hin, U0b, Gpre + (size_t)t * BATCH * G4,
                                             bvec, cst, hout, H0 + (size_t)t * BATCH * HID);
  }

  // 4) layer 1
  k_gemm_bf16<<<dim3(G4 / 64, ROWS / 64), 128, 0, stream>>>(H0, W1b, Gpre, nullptr, G4, HID);
  (void)hipMemsetAsync(cst, 0, (size_t)BATCH * HID * 4, stream);
  (void)hipMemsetAsync(hpp, 0, (size_t)2 * BATCH * HID * 2, stream);
  for (int t = 0; t < SEQL; ++t) {
    const __bf16* hin = hpp + (size_t)(t & 1) * BATCH * HID;
    __bf16* hout      = hpp + (size_t)((t + 1) & 1) * BATCH * HID;
    k_lstm_step<<<HID / 16, 64, 0, stream>>>(hin, U1b, Gpre + (size_t)t * BATCH * G4,
                                             bvec, cst, hout, H1 + (size_t)t * BATCH * HID);
  }

  // 5) decoder: out[4096, 32000] = H1 @ dec_w^T + dec_b  (dominant GEMM)
  k_gemm_bf16<<<dim3(NVOC / 64, ROWS / 64), 128, 0, stream>>>(H1, Db, out, dec_b, NVOC, HID);
}
